// CrossModalMambaModel_21955872817289
// MI455X (gfx1250) — compile-verified, matrix operand order checked by reading, not run on hardware
//
#include <hip/hip_runtime.h>
#include <hip/hip_bf16.h>

// ---------------------------------------------------------------------------
// CDNA5 (gfx1250) implementation: cross-modal attention + Mamba block.
// All GEMMs run through one register-blocked bf16 WMMA kernel:
// each wave computes a 64x32 macro-tile = 4x2 grid of v_wmma_f32_16x16x32_bf16
// accumulators, amortizing fragment loads 2.7x vs. naive per-tile waves.
// ---------------------------------------------------------------------------

typedef __attribute__((ext_vector_type(16))) __bf16 v16bf;
typedef __attribute__((ext_vector_type(8)))  float  v8f;

#define B_   8
#define L_   2048
#define AD_  512
#define VD_  512
#define H_   256
#define DIN_ 512
#define DST_ 16
#define DCONV_ 4
#define DTR_ 16
#define NCLS_ 8
#define BL_  (B_ * L_)

__device__ __forceinline__ __bf16 f2bf(float f) {
    union { float f; unsigned u; } v; v.f = f;
    unsigned r = v.u + 0x7FFFu + ((v.u >> 16) & 1u);   // round-to-nearest-even
    unsigned short h = (unsigned short)(r >> 16);
    return __builtin_bit_cast(__bf16, h);
}

__device__ __forceinline__ float silu(float v) {
    return v * (1.0f / (1.0f + __expf(-v)));
}

// ------------------------- fp32 -> bf16 convert -----------------------------
__global__ void f2b_kernel(const float* __restrict__ in, __bf16* __restrict__ out, long n) {
    long i = (long)blockIdx.x * blockDim.x + threadIdx.x;
    if (i < n) out[i] = f2bf(in[i]);
}

// ------------------- register-blocked WMMA bf16 GEMM ------------------------
// C[M,N] = scale * (A[M,K] @ W[N,K]^T) + bias   (optionally batched)
// Wave macro-tile: 64 rows x 32 cols = 4 m-tiles x 2 n-tiles of 16x16.
// Per K-step: 4 A-frags + 2 B-frags feed 8 chained WMMAs.
// A fragment: lane l holds row (l&15), K = k0 + ((l>>4)<<4) .. +15  (contig)
// B fragment: lane l holds col (l&15) of W, same contiguous K slice  (contig)
// C fragment: lane l holds rows ((l>>4)<<3)+0..7 at col (l&15)
__global__ void gemm_bf16_wmma(const __bf16* __restrict__ A,
                               const __bf16* __restrict__ W,
                               const float* __restrict__ bias,
                               float* __restrict__ Cf,
                               __bf16* __restrict__ Cb,
                               int M, int N, int K,
                               float scale, int transB16, int Lrows,
                               long sA, long sW, long sCf, long sCb) {
    const int wave = threadIdx.x >> 5;
    const int lane = threadIdx.x & 31;
    const int mt0 = (blockIdx.y * 4 + wave) * 4;   // first 16-row tile (4 per wave)
    if (mt0 * 16 >= M) return;
    const int nt0 = blockIdx.x * 2;                // first 16-col tile (2 per wave)
    const long b = blockIdx.z;
    const int ntile_cnt = N >> 4;

    const __bf16* Ab = A + b * sA;
    const __bf16* Wb = W + b * sW;

    const int r  = lane & 15;          // A-row / W-row(=col of B) within tile
    const int kh = (lane >> 4) << 4;   // 0 or 16: K half owned by this lane

    const __bf16* ap[4];
#pragma unroll
    for (int i = 0; i < 4; ++i)
        ap[i] = Ab + ((long)(mt0 + i) * 16 + r) * K + kh;
    const __bf16* wp[2];
#pragma unroll
    for (int j = 0; j < 2; ++j) {
        int nt = nt0 + j;
        if (nt >= ntile_cnt) nt = ntile_cnt - 1;   // clamp (ragged N); stores guarded
        wp[j] = Wb + ((long)nt * 16 + r) * K + kh;
    }

    v8f acc[4][2];
#pragma unroll
    for (int i = 0; i < 4; ++i)
#pragma unroll
        for (int j = 0; j < 2; ++j) acc[i][j] = {};

    for (int k0 = 0; k0 < K; k0 += 32) {
        v16bf af[4], wf[2];
#pragma unroll
        for (int i = 0; i < 4; ++i) {
            __builtin_prefetch(ap[i] + k0 + 256, 0, 1);
            af[i] = *(const v16bf*)(ap[i] + k0);
        }
#pragma unroll
        for (int j = 0; j < 2; ++j) {
            __builtin_prefetch(wp[j] + k0 + 256, 0, 1);
            wf[j] = *(const v16bf*)(wp[j] + k0);
        }
#pragma unroll
        for (int i = 0; i < 4; ++i)
#pragma unroll
            for (int j = 0; j < 2; ++j)
                acc[i][j] = __builtin_amdgcn_wmma_f32_16x16x32_bf16(
                    false, af[i], false, wf[j], (short)0, acc[i][j], false, false);
    }

    const int mrow = ((lane >> 4) << 3);
#pragma unroll
    for (int j = 0; j < 2; ++j) {
        const int nt = nt0 + j;
        if (nt >= ntile_cnt) continue;
        const int n = nt * 16 + r;
        const float bv = bias ? bias[n] : 0.0f;
#pragma unroll
        for (int i = 0; i < 4; ++i) {
            const int mbase = (mt0 + i) * 16 + mrow;
#pragma unroll
            for (int q = 0; q < 8; ++q) {
                const int m = mbase + q;
                float v = acc[i][j][q] * scale + bv;
                if (Cf) Cf[b * sCf + (long)m * N + n] = v;
                if (Cb) {
                    if (transB16) {    // store output transposed (per batch of Lrows)
                        int bb = m / Lrows, lm = m - bb * Lrows;
                        Cb[((long)bb * N + n) * (long)Lrows + lm] = f2bf(v);
                    } else {
                        Cb[b * sCb + (long)m * N + n] = f2bf(v);
                    }
                }
            }
        }
    }
}

// ------------------------- row softmax (f32 -> bf16) ------------------------
__global__ void softmax_rows(const float* __restrict__ S, __bf16* __restrict__ P, int ncols) {
    const long row = blockIdx.x;
    const float* s = S + row * (long)ncols;
    __bf16* p = P + row * (long)ncols;
    __shared__ float red[256];
    const int tid = threadIdx.x;

    float mx = -1e30f;
    for (int c = tid; c < ncols; c += 256) mx = fmaxf(mx, s[c]);
    red[tid] = mx; __syncthreads();
    for (int off = 128; off > 0; off >>= 1) {
        if (tid < off) red[tid] = fmaxf(red[tid], red[tid + off]);
        __syncthreads();
    }
    mx = red[0]; __syncthreads();

    float sum = 0.0f;
    for (int c = tid; c < ncols; c += 256) sum += __expf(s[c] - mx);
    red[tid] = sum; __syncthreads();
    for (int off = 128; off > 0; off >>= 1) {
        if (tid < off) red[tid] += red[tid + off];
        __syncthreads();
    }
    const float inv = 1.0f / red[0];
    for (int c = tid; c < ncols; c += 256) p[c] = f2bf(__expf(s[c] - mx) * inv);
}

// ------------------- depthwise causal conv (k=4) + bias + SiLU --------------
__global__ void conv_silu_kernel(const float* __restrict__ xz,
                                 const float* __restrict__ cw,
                                 const float* __restrict__ cb,
                                 __bf16* __restrict__ x16,
                                 float* __restrict__ xf) {
    long idx = (long)blockIdx.x * 256 + threadIdx.x;     // over B*L*DIN
    if (idx >= (long)BL_ * DIN_) return;
    int d = (int)(idx & (DIN_ - 1));
    long bl = idx >> 9;
    int l = (int)(bl & (L_ - 1));
    long b = bl >> 11;
    float acc = cb[d];
#pragma unroll
    for (int j = 0; j < DCONV_; ++j) {
        int ls = l - (DCONV_ - 1) + j;
        if (ls >= 0) acc += xz[(b * L_ + ls) * (2 * DIN_) + d] * cw[d * DCONV_ + j];
    }
    float v = silu(acc);
    x16[idx] = f2bf(v);
    xf[idx]  = v;
}

// ----------------- dt = softplus(dtc @ dt_proj_w^T + dt_proj_b) -------------
__global__ void dt_kernel(const float* __restrict__ dbc,
                          const float* __restrict__ dtw,
                          const float* __restrict__ dtb,
                          float* __restrict__ dtv) {
    long idx = (long)blockIdx.x * 256 + threadIdx.x;     // over B*L*DIN
    if (idx >= (long)BL_ * DIN_) return;
    int d = (int)(idx & (DIN_ - 1));
    long bl = idx >> 9;
    const float* r = dbc + bl * (DTR_ + 2 * DST_);
    const float* w = dtw + (long)d * DTR_;
    float acc = dtb[d];
#pragma unroll
    for (int i = 0; i < DTR_; ++i) acc += r[i] * w[i];
    dtv[idx] = (acc > 20.0f) ? acc : log1pf(__expf(acc));
}

// ------------------- selective scan + skip + gate (SiLU(z)) -----------------
// one thread per (batch, channel); h[16] in registers; B_t/C_t via LDS.
__global__ void mamba_scan_kernel(const float* __restrict__ dtv,
                                  const float* __restrict__ dbc,
                                  const float* __restrict__ xf,
                                  const float* __restrict__ xz,
                                  const float* __restrict__ A_log,
                                  const float* __restrict__ Dp,
                                  __bf16* __restrict__ y16) {
    const int b = blockIdx.x >> 1;
    const int d = ((blockIdx.x & 1) << 8) + threadIdx.x;
    float A[DST_];
#pragma unroll
    for (int s = 0; s < DST_; ++s) A[s] = -__expf(A_log[(long)d * DST_ + s]);
    const float Dd = Dp[d];
    float h[DST_];
#pragma unroll
    for (int s = 0; s < DST_; ++s) h[s] = 0.0f;

    __shared__ float Bs[DST_], Cs[DST_];
    const long base = (long)b * L_;
    for (int l = 0; l < L_; ++l) {
        const long blk = base + l;
        if (threadIdx.x < 32) {
            const float* row = dbc + blk * (DTR_ + 2 * DST_) + DTR_;
            if (threadIdx.x < 16) Bs[threadIdx.x] = row[threadIdx.x];
            else                  Cs[threadIdx.x - 16] = row[DST_ + threadIdx.x - 16];
        }
        __syncthreads();
        const float dt_t = dtv[blk * DIN_ + d];
        const float x_t  = xf[blk * DIN_ + d];
        const float dx   = dt_t * x_t;
        float y = 0.0f;
#pragma unroll
        for (int s = 0; s < DST_; ++s) {
            const float dA = __expf(dt_t * A[s]);
            h[s] = dA * h[s] + dx * Bs[s];
            y += h[s] * Cs[s];
        }
        y += x_t * Dd;
        const float z = xz[blk * (2 * DIN_) + DIN_ + d];
        y *= silu(z);
        y16[blk * DIN_ + d] = f2bf(y);
        __syncthreads();
    }
}

// ------------------------- mean pool over L ---------------------------------
__global__ void pool_kernel(const float* __restrict__ mo, float* __restrict__ pooled) {
    int idx = blockIdx.x * 256 + threadIdx.x;   // over B*H
    if (idx >= B_ * H_) return;
    int b = idx >> 8, hn = idx & (H_ - 1);
    float acc = 0.0f;
    const float* p = mo + (long)b * L_ * H_ + hn;
    for (int l = 0; l < L_; ++l) acc += p[(long)l * H_];
    pooled[idx] = acc * (1.0f / L_);
}

// ------------------------- classifier + softmax -----------------------------
__global__ void cls_kernel(const float* __restrict__ pooled,
                           const float* __restrict__ cw,
                           const float* __restrict__ cb,
                           float* __restrict__ out) {
    __shared__ float lg[B_ * NCLS_];
    const int t = threadIdx.x;              // 64 threads: b*8+c
    const int b = t >> 3, c = t & 7;
    float acc = cb[c];
    const float* pv = pooled + (long)b * H_;
    const float* wv = cw + (long)c * H_;
    for (int i = 0; i < H_; ++i) acc += pv[i] * wv[i];
    lg[t] = acc;
    out[t] = acc;                           // logits
    __syncthreads();
    float mx = lg[b * 8];
#pragma unroll
    for (int i = 1; i < NCLS_; ++i) mx = fmaxf(mx, lg[b * 8 + i]);
    float den = 0.0f;
#pragma unroll
    for (int i = 0; i < NCLS_; ++i) den += __expf(lg[b * 8 + i] - mx);
    out[B_ * NCLS_ + t] = __expf(acc - mx) / den;   // preds
}

// ---------------------------------------------------------------------------
extern "C" void kernel_launch(void* const* d_in, const int* in_sizes, int n_in,
                              void* d_out, int out_size, void* d_ws, size_t ws_size,
                              hipStream_t stream) {
    const float* audio   = (const float*)d_in[0];
    const float* visual  = (const float*)d_in[1];
    const float* audio_w = (const float*)d_in[2];
    const float* audio_b = (const float*)d_in[3];
    const float* vis_w   = (const float*)d_in[4];
    const float* vis_b   = (const float*)d_in[5];
    const float* q_w = (const float*)d_in[6];  const float* q_b = (const float*)d_in[7];
    const float* k_w = (const float*)d_in[8];  const float* k_b = (const float*)d_in[9];
    const float* v_w = (const float*)d_in[10]; const float* v_b = (const float*)d_in[11];
    const float* in_proj_w = (const float*)d_in[12];
    const float* conv_w = (const float*)d_in[13];
    const float* conv_b = (const float*)d_in[14];
    const float* x_proj_w = (const float*)d_in[15];
    const float* dt_proj_w = (const float*)d_in[16];
    const float* dt_proj_b = (const float*)d_in[17];
    const float* A_log = (const float*)d_in[18];
    const float* Dvec  = (const float*)d_in[19];
    const float* out_proj_w = (const float*)d_in[20];
    const float* cls_w = (const float*)d_in[21];
    const float* cls_b = (const float*)d_in[22];
    float* out = (float*)d_out;

    // ---- workspace carve ----
    char* base = (char*)d_ws;
    size_t off = 0;
    auto carve = [&](size_t bytes) -> char* {
        char* p = base + off;
        off = (off + bytes + 255) & ~(size_t)255;
        return p;
    };
    __bf16* ab16   = (__bf16*)carve((size_t)BL_ * AD_ * 2);
    __bf16* vb16   = (__bf16*)carve((size_t)BL_ * VD_ * 2);
    __bf16* wa16   = (__bf16*)carve((size_t)H_ * AD_ * 2);
    __bf16* wv16   = (__bf16*)carve((size_t)H_ * VD_ * 2);
    __bf16* wq16   = (__bf16*)carve((size_t)H_ * H_ * 2);
    __bf16* wk16   = (__bf16*)carve((size_t)H_ * H_ * 2);
    __bf16* wvp16  = (__bf16*)carve((size_t)H_ * H_ * 2);
    __bf16* wip16  = (__bf16*)carve((size_t)2 * DIN_ * H_ * 2);
    __bf16* wxp16  = (__bf16*)carve((size_t)(DTR_ + 2 * DST_) * DIN_ * 2);
    __bf16* wop16  = (__bf16*)carve((size_t)H_ * DIN_ * 2);
    __bf16* ah16   = (__bf16*)carve((size_t)BL_ * H_ * 2);
    __bf16* vh16   = (__bf16*)carve((size_t)BL_ * H_ * 2);
    __bf16* Q16    = (__bf16*)carve((size_t)BL_ * H_ * 2);
    __bf16* K16    = (__bf16*)carve((size_t)BL_ * H_ * 2);
    __bf16* Vt16   = (__bf16*)carve((size_t)BL_ * H_ * 2);   // B x H x L (transposed)
    float*  scores = (float*) carve((size_t)B_ * L_ * L_ * 4);
    __bf16* attn16 = (__bf16*)carve((size_t)B_ * L_ * L_ * 2);
    __bf16* fus16  = (__bf16*)carve((size_t)BL_ * H_ * 2);
    float*  xz     = (float*) carve((size_t)BL_ * 2 * DIN_ * 4);
    __bf16* x16    = (__bf16*)carve((size_t)BL_ * DIN_ * 2);
    float*  xf     = (float*) carve((size_t)BL_ * DIN_ * 4);
    float*  dbc    = (float*) carve((size_t)BL_ * (DTR_ + 2 * DST_) * 4);
    float*  dtbuf  = (float*) carve((size_t)BL_ * DIN_ * 4);
    __bf16* y16    = (__bf16*)carve((size_t)BL_ * DIN_ * 2);
    float*  mo     = (float*) carve((size_t)BL_ * H_ * 4);
    float*  pooled = (float*) carve((size_t)B_ * H_ * 4);
    (void)ws_size; (void)in_sizes; (void)n_in; (void)out_size;

    auto cvt = [&](const float* src, __bf16* dst, long n) {
        f2b_kernel<<<dim3((unsigned)((n + 255) / 256)), dim3(256), 0, stream>>>(src, dst, n);
    };
    // activations + weights to bf16
    cvt(audio,  ab16, (long)BL_ * AD_);
    cvt(visual, vb16, (long)BL_ * VD_);
    cvt(audio_w, wa16, (long)H_ * AD_);
    cvt(vis_w,   wv16, (long)H_ * VD_);
    cvt(q_w, wq16, (long)H_ * H_);
    cvt(k_w, wk16, (long)H_ * H_);
    cvt(v_w, wvp16, (long)H_ * H_);
    cvt(in_proj_w, wip16, (long)2 * DIN_ * H_);
    cvt(x_proj_w,  wxp16, (long)(DTR_ + 2 * DST_) * DIN_);
    cvt(out_proj_w, wop16, (long)H_ * DIN_);

    // grid: x over 32-col macro-tiles, y over 256-row blocks (4 waves x 64 rows)
    auto gemm = [&](const __bf16* A, const __bf16* W, const float* bias,
                    float* Cf, __bf16* Cb, int M, int N, int K,
                    float scale, int trans, int batch,
                    long sA, long sW, long sCf, long sCb) {
        dim3 grid((unsigned)((N + 31) / 32), (unsigned)((M + 255) / 256), (unsigned)batch);
        gemm_bf16_wmma<<<grid, dim3(128), 0, stream>>>(
            A, W, bias, Cf, Cb, M, N, K, scale, trans, L_, sA, sW, sCf, sCb);
    };

    // modality projections (f32 in -> bf16 hidden)
    gemm(ab16, wa16, audio_b, nullptr, ah16, BL_, H_, AD_, 1.0f, 0, 1, 0, 0, 0, 0);
    gemm(vb16, wv16, vis_b,   nullptr, vh16, BL_, H_, VD_, 1.0f, 0, 1, 0, 0, 0, 0);
    // Q / K / V (V stored transposed per batch for the PV GEMM)
    gemm(ah16, wq16,  q_b, nullptr, Q16, BL_, H_, H_, 1.0f, 0, 1, 0, 0, 0, 0);
    gemm(vh16, wk16,  k_b, nullptr, K16, BL_, H_, H_, 1.0f, 0, 1, 0, 0, 0, 0);
    gemm(vh16, wvp16, v_b, nullptr, Vt16, BL_, H_, H_, 1.0f, 1, 1, 0, 0, 0, 0);
    // scores = Q K^T / sqrt(H)   (batched; 134MB fits in 192MB L2)
    gemm(Q16, K16, nullptr, scores, nullptr, L_, L_, H_, 0.0625f, 0, B_,
         (long)L_ * H_, (long)L_ * H_, (long)L_ * L_, 0);
    // softmax rows -> bf16 probs
    softmax_rows<<<dim3(BL_), dim3(256), 0, stream>>>(scores, attn16, L_);
    // fused = P @ V   (contract over L via transposed V)
    gemm(attn16, Vt16, nullptr, nullptr, fus16, L_, H_, L_, 1.0f, 0, B_,
         (long)L_ * L_, (long)H_ * L_, 0, (long)L_ * H_);
    // mamba in_proj -> xz (f32, split x|z)
    gemm(fus16, wip16, nullptr, xz, nullptr, BL_, 2 * DIN_, H_, 1.0f, 0, 1, 0, 0, 0, 0);
    // depthwise causal conv + SiLU
    conv_silu_kernel<<<dim3((BL_ * DIN_) / 256), dim3(256), 0, stream>>>(
        xz, conv_w, conv_b, x16, xf);
    // x_proj -> (dt | B | C)
    gemm(x16, wxp16, nullptr, dbc, nullptr, BL_, DTR_ + 2 * DST_, DIN_, 1.0f, 0, 1, 0, 0, 0, 0);
    // dt = softplus(dt_proj)
    dt_kernel<<<dim3((BL_ * DIN_) / 256), dim3(256), 0, stream>>>(
        dbc, dt_proj_w, dt_proj_b, dtbuf);
    // selective scan + skip + gate
    mamba_scan_kernel<<<dim3(B_ * (DIN_ / 256)), dim3(256), 0, stream>>>(
        dtbuf, dbc, xf, xz, A_log, Dvec, y16);
    // out_proj
    gemm(y16, wop16, nullptr, mo, nullptr, BL_, H_, DIN_, 1.0f, 0, 1, 0, 0, 0, 0);
    // mean pool + classifier + softmax
    pool_kernel<<<dim3((B_ * H_ + 255) / 256), dim3(256), 0, stream>>>(mo, pooled);
    cls_kernel<<<dim3(1), dim3(B_ * NCLS_), 0, stream>>>(pooled, cls_w, cls_b, out);
}
